// JitteredFeedbackCell_65146063946217
// MI455X (gfx1250) — matrix operand order repaired and checked by reading, not compiled, not run on hardware
//
#include <hip/hip_runtime.h>
#include <math.h>

typedef _Float16 h16;
typedef _Float16 v8h  __attribute__((ext_vector_type(8)));
typedef _Float16 v16h __attribute__((ext_vector_type(16)));
typedef float    v8f  __attribute__((ext_vector_type(8)));

#define UNITS   256
#define D_IN    256
#define BATCH   65536
#define MTILE   32
#define APITCH  264     // pitch (elements) for 256-wide row tiles (+8 pad vs LDS banks)
#define KSLICE  64      // K rows of stacked weights per slice
#define NSLICES 8       // 512 / 64
#define WPITCH  72      // halves per weight row in packed layout (64 + 8 pad)
#define SLICE_BYTES (UNITS*WPITCH*2)   // 36864 B per slice (=9 b128 per thread @256 thr)
#define WPACK_BYTES (NSLICES*SLICE_BYTES)

// ---- dynamic LDS carve-up (bytes) ----
#define OFF_H32   0
#define SZ_H32    (MTILE*APITCH*4)          // prev_h tile, f32 (roll, A-frag src, epilogue)
#define OFF_G16   (OFF_H32 + SZ_H32)
#define SZ_G16    (MTILE*APITCH*2)          // new_G tile, f16 (stats + tanh input)
#define OFF_AIN   (OFF_G16 + SZ_G16)
#define SZ_AIN    (MTILE*APITCH*2)          // inputs tile, f16
#define OFF_W     (OFF_AIN + SZ_AIN)
#define SZ_W      (2*SLICE_BYTES)           // double-buffered weight slice, f16 [n][k]
#define OFF_BIAS  (OFF_W + SZ_W)
#define SZ_BIAS   (UNITS*4)
#define OFF_PART  (OFF_BIAS + SZ_BIAS)
#define SZ_PART   (MTILE*8*3*4)
#define OFF_STAT  (OFF_PART + SZ_PART)
#define SZ_STAT   (MTILE*4*4)
#define LDS_TOTAL (OFF_STAT + SZ_STAT)      // 145920 B < 160 KB -> 2 WGs/WGP

#define PHASE_STEP 0.25132741228718345f     // 2*pi/25

// ============ weight pre-pack: f32 [k][n] -> f16 [slice][n][WPITCH] ============
__global__ void __launch_bounds__(256)
pack_weights_kernel(const float* __restrict__ w_in, const float* __restrict__ w_rec,
                    unsigned* __restrict__ wpack32)
{
    const int s = blockIdx.x;        // 0..7 slice of stacked [w_in; w_rec]
    const int n = threadIdx.x;       // 0..255 output column
    const float* wsrc = (s < 4) ? (w_in + (size_t)s * KSLICE * UNITS)
                                : (w_rec + (size_t)(s - 4) * KSLICE * UNITS);
    unsigned* dst = wpack32 + ((size_t)s * UNITS + n) * (WPITCH/2);
    #pragma unroll 4
    for (int kp = 0; kp < 32; ++kp) {
        union { h16 h[2]; unsigned u; } cv;
        cv.h[0] = (h16)wsrc[(2*kp + 0)*UNITS + n];
        cv.h[1] = (h16)wsrc[(2*kp + 1)*UNITS + n];
        dst[kp] = cv.u;
    }
    dst[32] = 0u; dst[33] = 0u; dst[34] = 0u; dst[35] = 0u;   // pad halves 64..71
}

// ============ fused cell kernel ============
__device__ __forceinline__ void issue_slice_async(const char* wpack, int s, int buf, int tid)
{
    // each thread DMA-copies 9 x 16B; whole WG covers one 36864B slice
    const unsigned lds0 = __builtin_amdgcn_groupstaticsize()
                        + (unsigned)(OFF_W + buf*SLICE_BYTES + tid*16);
    const unsigned goff = (unsigned)(tid*16);
    const char* base = wpack + (size_t)s * SLICE_BYTES;
    #pragma unroll
    for (int j = 0; j < 9; ++j) {
        unsigned l = lds0 + (unsigned)(j*4096);
        asm volatile("global_load_async_to_lds_b128 %0, %1, %2 offset:%3"
                     :: "v"(l), "v"(goff), "s"(base), "i"(j*4096)
                     : "memory");
    }
}

__global__ void __launch_bounds__(256)
jittered_cell_kernel(const float* __restrict__ inputs,
                     const float* __restrict__ prev_h,
                     const float* __restrict__ prev_G,
                     const float* __restrict__ prev_phase,
                     const float* __restrict__ bias,
                     const char*  __restrict__ wpack,
                     float* __restrict__ out_h,
                     float* __restrict__ out_G,
                     float* __restrict__ out_phase)
{
    extern __shared__ char smem[];
    float* s_h32  = (float*)(smem + OFF_H32);
    h16*   s_g16  = (h16*)  (smem + OFF_G16);
    h16*   s_ain  = (h16*)  (smem + OFF_AIN);
    float* s_bias = (float*)(smem + OFF_BIAS);
    float* s_part = (float*)(smem + OFF_PART);
    float* s_stat = (float*)(smem + OFF_STAT);

    const int tid  = threadIdx.x;
    const int lane = tid & 31;
    const int wave = tid >> 5;
    const int row0 = blockIdx.x * MTILE;

    // kick off slice-0 weight DMA immediately; it overlaps all staging below
    issue_slice_async(wpack, 0, 0, tid);

    // ---- stage prev_h (f32) and inputs (f16); NT streaming reads ----
    #pragma unroll 4
    for (int i = 0; i < MTILE; ++i) {
        const int r = row0 + i;
        s_h32[i*APITCH + tid] = __builtin_nontemporal_load(&prev_h[r*UNITS + tid]);
        s_ain[i*APITCH + tid] = (h16)__builtin_nontemporal_load(&inputs[r*UNITS + tid]);
    }
    s_bias[tid] = bias[tid];
    __syncthreads();

    // ---- new_G = 0.9*prev_G + 0.1*roll(prev_h,128); f32 out, f16 kept in LDS ----
    #pragma unroll 4
    for (int i = 0; i < MTILE; ++i) {
        const int r = row0 + i;
        const float g = 0.9f * __builtin_nontemporal_load(&prev_G[r*UNITS + tid])
                      + 0.1f * s_h32[i*APITCH + ((tid + 128) & 255)];
        s_g16[i*APITCH + tid] = (h16)g;
        __builtin_nontemporal_store(g, &out_G[r*UNITS + tid]);
    }
    __syncthreads();

    // ---- per-row partial reductions: 8 threads x 32 elems per row ----
    {
        const int r   = tid >> 3;
        const int seg = tid & 7;
        float sg = 0.f, sg2 = 0.f, sh = 0.f;
        const int base = r*APITCH + seg*32;
        #pragma unroll
        for (int j = 0; j < 32; ++j) {
            const float g = (float)s_g16[base + j];
            sg  += g;
            sg2 += g * g;
            sh  += s_h32[base + j];    // roll preserves the row sum
        }
        s_part[(r*8 + seg)*3 + 0] = sg;
        s_part[(r*8 + seg)*3 + 1] = sg2;
        s_part[(r*8 + seg)*3 + 2] = sh;
    }
    __syncthreads();

    if (tid < MTILE) {
        float sg = 0.f, sg2 = 0.f, sh = 0.f;
        #pragma unroll
        for (int q = 0; q < 8; ++q) {
            sg  += s_part[(tid*8 + q)*3 + 0];
            sg2 += s_part[(tid*8 + q)*3 + 1];
            sh  += s_part[(tid*8 + q)*3 + 2];
        }
        const float mean = sg * (1.0f/UNITS);
        float var = sg2 * (1.0f/UNITS) - mean*mean;
        var = fmaxf(var, 0.0f);
        const float inv = 1.0f / (sqrtf(var) + 1e-6f);
        const float ph  = prev_phase[row0 + tid] + PHASE_STEP;
        out_phase[row0 + tid] = ph;
        const float combined = sinf(ph) + 0.05f * (sh * (1.0f/UNITS) - 0.1f);
        s_stat[tid*4 + 0] = mean;
        s_stat[tid*4 + 1] = inv;
        s_stat[tid*4 + 2] = 0.5f * combined;
    }

    // ---- GEMM: z = [inputs | prev_h] @ [w_in ; w_rec], unified K = 512 ----
    const int mb    = wave & 1;
    const int nb    = wave >> 1;
    const int mrow  = mb*16 + (lane & 15);
    const int khalf = lane >> 4;

    v8f acc[4] = {};

    auto compute_slice = [&](int s, int buf) {
        const h16*  s_w  = (const h16*)(smem + OFF_W + buf*SLICE_BYTES);
        const bool  useH = (s >= 4);
        const int   acol0 = (s & 3) * KSLICE;
        #pragma unroll
        for (int k0 = 0; k0 < KSLICE; k0 += 32) {
            // A fragment (ISA 16-bit A 16x32 lane map)
            v16h a;
            const int c = acol0 + k0 + 8*khalf;
            if (useH) {
                const float4* ap = (const float4*)&s_h32[mrow*APITCH + c];
                const float4 f0 = ap[0];
                const float4 f1 = ap[1];
                const float4 f2 = *(const float4*)&s_h32[mrow*APITCH + c + 16];
                const float4 f3 = *(const float4*)&s_h32[mrow*APITCH + c + 20];
                a[0]=(h16)f0.x; a[1]=(h16)f0.y; a[2]=(h16)f0.z; a[3]=(h16)f0.w;
                a[4]=(h16)f1.x; a[5]=(h16)f1.y; a[6]=(h16)f1.z; a[7]=(h16)f1.w;
                a[8]=(h16)f2.x; a[9]=(h16)f2.y; a[10]=(h16)f2.z; a[11]=(h16)f2.w;
                a[12]=(h16)f3.x; a[13]=(h16)f3.y; a[14]=(h16)f3.z; a[15]=(h16)f3.w;
            } else {
                const h16* ap = &s_ain[mrow*APITCH + c];
                const v8h lo = *(const v8h*)ap;
                const v8h hi = *(const v8h*)(ap + 16);
                #pragma unroll
                for (int j = 0; j < 8; ++j) { a[j] = lo[j]; a[8+j] = hi[j]; }
            }
            // load ALL B fragments first, then issue the 4 WMMAs (overlap ds/wmma)
            v16h b[4];
            #pragma unroll
            for (int t = 0; t < 4; ++t) {
                const int n = nb*64 + t*16 + (lane & 15);
                const h16* bp = &s_w[n*WPITCH + k0 + 8*khalf];
                const v8h b0 = *(const v8h*)bp;
                const v8h b1 = *(const v8h*)(bp + 16);
                #pragma unroll
                for (int j = 0; j < 8; ++j) { b[t][j] = b0[j]; b[t][8+j] = b1[j]; }
            }
            #pragma unroll
            for (int t = 0; t < 4; ++t) {
                acc[t] = __builtin_amdgcn_wmma_f32_16x16x32_f16(
                    false, a, false, b[t], (short)0, acc[t], false, false);
            }
        }
    };

    // software pipeline: DMA slice s+1 while computing slice s
    for (int s = 0; s < NSLICES - 1; ++s) {
        issue_slice_async(wpack, s + 1, (s + 1) & 1, tid);
        asm volatile("s_wait_asynccnt 0x9" ::: "memory");   // my slice-s copies done
        __syncthreads();                                     // everyone's slice-s copies done
        compute_slice(s, s & 1);
        __syncthreads();                                     // buf[s&1] free for iter s+1
    }
    asm volatile("s_wait_asynccnt 0x0" ::: "memory");
    __syncthreads();
    compute_slice(NSLICES - 1, (NSLICES - 1) & 1);

    // ---- fused epilogue: field_effect, elu, inertia, clip ----
    #pragma unroll
    for (int t = 0; t < 4; ++t) {
        const int   n    = nb*64 + t*16 + (lane & 15);
        const float bval = s_bias[n];
        #pragma unroll
        for (int i = 0; i < 8; ++i) {
            const int   m    = mb*16 + i + 8*khalf;          // C-matrix lane map
            const float mean = s_stat[m*4 + 0];
            const float inv  = s_stat[m*4 + 1];
            const float comb = s_stat[m*4 + 2];
            const float g    = (float)s_g16[m*APITCH + n];
            const float fe   = 1.0f + comb * tanhf((g - mean) * inv);
            const float z    = (acc[t][i] + bval) * fe;
            const float e    = (z > 0.0f) ? z : expm1f(z);
            float h = 0.9f * s_h32[m*APITCH + n] + 0.1f * e;
            h = fminf(20.0f, fmaxf(-20.0f, h));
            __builtin_nontemporal_store(h, &out_h[(row0 + m)*UNITS + n]);
        }
    }
}

extern "C" void kernel_launch(void* const* d_in, const int* in_sizes, int n_in,
                              void* d_out, int out_size, void* d_ws, size_t ws_size,
                              hipStream_t stream) {
    const float* inputs     = (const float*)d_in[0];
    const float* prev_h     = (const float*)d_in[1];
    const float* prev_G     = (const float*)d_in[2];
    const float* prev_phase = (const float*)d_in[3];
    const float* w_in       = (const float*)d_in[4];
    const float* w_rec      = (const float*)d_in[5];
    const float* bias       = (const float*)d_in[6];

    float* out_h     = (float*)d_out;
    float* out_G     = out_h + (size_t)BATCH * UNITS;
    float* out_phase = out_G + (size_t)BATCH * UNITS;

    // d_ws: packed f16 transposed weights, WPACK_BYTES = 294912 B
    unsigned* wpack32 = (unsigned*)d_ws;

    pack_weights_kernel<<<dim3(NSLICES), dim3(256), 0, stream>>>(w_in, w_rec, wpack32);

    jittered_cell_kernel<<<dim3(BATCH / MTILE), dim3(256), LDS_TOTAL, stream>>>(
        inputs, prev_h, prev_G, prev_phase, bias, (const char*)d_ws,
        out_h, out_G, out_phase);
}
